// ConvPMF_62929860821268
// MI455X (gfx1250) — matrix operand order repaired and checked by reading, not compiled, not run on hardware
//
#include <hip/hip_runtime.h>
#include <math.h>

// Problem constants (match reference).
#define BATCH 128
#define RPD   10      // reviews per doc
#define WPW   256     // words per review
#define NFAC  32      // conv filters / latent factors
#define EDIM  128     // embedding dim
#define KWIN  5       // conv window
#define PAD   2       // zero word-rows each side for 'same' padding

#define EMB_WORDS (WPW + 2 * PAD)               // 260 word slots
#define EMB_BYTES (EMB_WORDS * EDIM * 2)        // 66560: bf16 [slot][dim]
#define WGT_ELEMS (KWIN * NFAC * EDIM)          // 20480 bf16 weights
#define WGT_BYTES (WGT_ELEMS * 2)               // 40960: bf16 [tap][f][d]
#define SMEM_BYTES (EMB_BYTES + WGT_BYTES)      // 107520 (< 320KB WGP LDS)

#define POOLED_BYTES (BATCH * RPD * NFAC * 4)   // 163840 in d_ws
// bf16 weights live in d_ws at byte offset POOLED_BYTES.

typedef __attribute__((ext_vector_type(16))) __bf16 v16bf;
typedef __attribute__((ext_vector_type(8)))  float  v8f;
typedef __attribute__((ext_vector_type(4)))  int    v4i;

typedef __attribute__((address_space(1))) void gvoid;   // global (AS1)
typedef __attribute__((address_space(3))) void lvoid;   // LDS    (AS3)
typedef __attribute__((address_space(1))) v4i  gv4i;
typedef __attribute__((address_space(3))) v4i  lv4i;

struct alignas(16) U128 { unsigned u0, u1, u2, u3; };

union Frag16 {
    v16bf v;
    U128  q[2];
    unsigned short s[16];
};

__device__ __forceinline__ unsigned short f2bf(float x) {
    return __builtin_bit_cast(unsigned short, (__bf16)x);   // hw cvt, RNE
}

#if __has_builtin(__builtin_amdgcn_global_load_async_to_lds_b128)
#define HAVE_ASYNC_LDS 1
__device__ __forceinline__ void async_copy_b128(const void* gsrc, void* ldst) {
    __builtin_amdgcn_global_load_async_to_lds_b128(
        (gv4i*)((gvoid*)gsrc), (lv4i*)((lvoid*)ldst), 0 /*imm off*/, 0 /*cpol*/);
}
#endif

// One-shot: convert conv weights f32 [32,128,5] -> bf16 transposed [tap][f][d].
__global__ __launch_bounds__(256)
void convpmf_wconv_kernel(const float* __restrict__ convw,
                          unsigned short* __restrict__ wbf) {
    int idx = blockIdx.x * 256 + threadIdx.x;   // [0, 20480)
    int k   = idx >> 12;                        // / (32*128)
    int rem = idx & 4095;
    int f   = rem >> 7;
    int d   = rem & 127;
    wbf[idx] = f2bf(convw[(f * EDIM + d) * KWIN + k]);
}

// One block (8 wave32) per review. Conv1d(same) as 5 shifted GEMMs via
// v_wmma_f32_16x16x32_bf16, then pooled[f] = 1 / sum_w exp(fm - rowmax).
__global__ __launch_bounds__(256)
void convpmf_review_kernel(const int*   __restrict__ docs,    // [B,R,W]
                           const float* __restrict__ embed,   // [VOCAB,128]
                           const unsigned short* __restrict__ wbf, // bf16 [tap][f][d]
                           float*       __restrict__ pooled)  // [B*R,32]
{
    __shared__ __attribute__((aligned(16))) unsigned char smem[SMEM_BYTES];
    unsigned short* embS = (unsigned short*)smem;               // bf16 [slot][dim]
    unsigned short* wS   = (unsigned short*)(smem + EMB_BYTES); // bf16 [tap][f][d]
    // Reuse of the embedding region after the GEMM phase:
    float* fmS   = (float*)smem;                  // f32 [32][256] = 32 KB
    float* redS  = (float*)(smem + 32768);        // 256 partials  = 1 KB
    float* rmaxS = (float*)(smem + 32768 + 1024); // 32 row maxima

    const int n    = blockIdx.x;            // review id in [0, B*R)
    const int tid  = threadIdx.x;
    const int lane = tid & 31;
    const int wave = tid >> 5;

    // ---- stage 1a: async-copy staged bf16 weights global -> LDS ----
    // 40960 B = 2560 x b128 chunks; 10 per thread. ASYNCcnt-tracked DMA,
    // no VGPR round-trip. Each wave drains its own loads before the barrier.
#ifdef HAVE_ASYNC_LDS
    {
        const unsigned char* src = (const unsigned char*)wbf;
        unsigned char*       dst = (unsigned char*)wS;
        #pragma unroll
        for (int it = 0; it < 10; ++it) {
            int byteOff = (it * 256 + tid) * 16;
            async_copy_b128(src + byteOff, dst + byteOff);
        }
    }
#else
    {
        const U128* src = (const U128*)wbf;
        U128*       dst = (U128*)wS;
        #pragma unroll
        for (int it = 0; it < 10; ++it) {
            int idx = it * 256 + tid;
            dst[idx] = src[idx];
        }
    }
#endif

    // ---- stage 1b: zero the pad rows (word slots 0,1 and 258,259) ----
    if (tid < 128) {
        ((unsigned*)smem)[tid] = 0u;                            // low pad: 512 B
        ((unsigned*)(smem + (WPW + PAD) * EDIM * 2))[tid] = 0u; // high pad: 512 B
    }

    // ---- stage 1c: gather embeddings -> LDS bf16 (slot = word + PAD) ----
    const float4* emb4 = (const float4*)embed;     // 32 float4 per row
    const int*    drow = docs + n * WPW;
    #pragma unroll
    for (int it = 0; it < 32; ++it) {
        int gidx = it * 256 + tid;                 // 8192 = 256 words * 32 f4
        int word = gidx >> 5;
        int c4   = gidx & 31;
        int widx = drow[word];
        float4 v = emb4[(size_t)widx * 32 + c4];
        unsigned lo = (unsigned)f2bf(v.x) | ((unsigned)f2bf(v.y) << 16);
        unsigned hi = (unsigned)f2bf(v.z) | ((unsigned)f2bf(v.w) << 16);
        unsigned* dst = (unsigned*)&embS[(word + PAD) * EDIM + c4 * 4];
        dst[0] = lo;
        dst[1] = hi;
    }

#ifdef HAVE_ASYNC_LDS
    asm volatile("s_wait_asynccnt 0x0" ::: "memory");  // drain this wave's DMA
#endif
    __syncthreads();

    // ---- stage 2: conv as 5 shifted GEMMs [32x128]x[128x256] via WMMA ----
    const int mtile = wave & 1;            // filters [0..15] or [16..31]
    const int nbase = (wave >> 1) * 4;     // this wave's 4 n-tiles (16 words each)

    const int fRow = mtile * 16 + (lane & 15);
    const int kb8  = (lane & 16) ? 8  : 0;   // A-frag K sub-base (ISA 16-bit A layout)
    const int kb16 = (lane & 16) ? 16 : 0;   // unused; B frags also use kb8 (same k order as A)

    // Per-lane base pointers; all tap/ks offsets become DS immediate offsets.
    // B: slot index = col + shift + PAD = col + tap  -> elem = col*128+kb8 + tap*128+k0
    const unsigned short* aBase = &wS[fRow * EDIM + kb8];
    const unsigned short* bBase[4];
    #pragma unroll
    for (int t = 0; t < 4; ++t) {
        int col = (nbase + t) * 16 + (lane & 15);
        bBase[t] = &embS[col * EDIM + kb8];
    }

    v8f acc[4];
    #pragma unroll
    for (int t = 0; t < 4; ++t)
        acc[t] = (v8f){0.f, 0.f, 0.f, 0.f, 0.f, 0.f, 0.f, 0.f};

    #pragma unroll
    for (int tap = 0; tap < KWIN; ++tap) {
        #pragma unroll
        for (int ks = 0; ks < 4; ++ks) {           // K = 128 in chunks of 32
            const int k0 = ks * 32;
            Frag16 a;                              // weights [16 x 32] bf16
            a.q[0] = *(const U128*)(aBase + tap * NFAC * EDIM + k0);
            a.q[1] = *(const U128*)(aBase + tap * NFAC * EDIM + k0 + 16);
            #pragma unroll
            for (int t = 0; t < 4; ++t) {
                Frag16 b;                          // shifted embeds [32 x 16] bf16
                b.q[0] = *(const U128*)(bBase[t] + tap * EDIM + k0);
                b.q[1] = *(const U128*)(bBase[t] + tap * EDIM + k0 + 16);
                acc[t] = __builtin_amdgcn_wmma_f32_16x16x32_bf16(
                    false, a.v, false, b.v, (short)0, acc[t], false, false);
            }
        }
    }
    __syncthreads();   // all reads of embS complete before fm overwrites it

    // ---- stage 3: spill fm tiles (f32) to LDS per ISA C/D layout ----
    const int fBase = mtile * 16 + ((lane & 16) ? 8 : 0);
    #pragma unroll
    for (int t = 0; t < 4; ++t) {
        int w = (nbase + t) * 16 + (lane & 15);
        #pragma unroll
        for (int r = 0; r < 8; ++r)
            fmS[(fBase + r) * WPW + w] = acc[t][r];
    }
    __syncthreads();

    // ---- stage 4: max(softmax) == 1 / sum_w exp(x - rowmax), 8 thr/row ----
    const int f   = tid >> 3;          // filter row 0..31
    const int sub = tid & 7;           // 8 partials per row, 32 w each
    const float* row = fmS + f * WPW + sub * 32;

    float pm = -INFINITY;
    #pragma unroll
    for (int w = 0; w < 32; ++w) pm = fmaxf(pm, row[w]);
    redS[tid] = pm;
    __syncthreads();

    if (sub == 0) {
        float m = redS[tid];
        #pragma unroll
        for (int j = 1; j < 8; ++j) m = fmaxf(m, redS[tid + j]);
        rmaxS[f] = m;
    }
    __syncthreads();

    float m = rmaxS[f];
    float ps = 0.f;
    #pragma unroll
    for (int w = 0; w < 32; ++w) ps += __expf(row[w] - m);
    __syncthreads();           // redS partial-max reads done before overwrite
    redS[tid] = ps;
    __syncthreads();

    if (sub == 0) {
        float s = redS[tid];
        #pragma unroll
        for (int j = 1; j < 8; ++j) s += redS[tid + j];
        pooled[n * NFAC + f] = 1.0f / s;
    }
}

// Ragged masked mean over reviews + user-factor dot + bias.
__global__ __launch_bounds__(128)
void convpmf_reduce_kernel(const float* __restrict__ pooled,        // [B*R,32]
                           const int*   __restrict__ user_indices,  // [B]
                           const int*   __restrict__ review_counts, // [B]
                           const float* __restrict__ w_user,        // [NUM_USER,32]
                           const float* __restrict__ bias,          // [1]
                           float*       __restrict__ out)           // [B]
{
    int b = blockIdx.x * blockDim.x + threadIdx.x;
    if (b >= BATCH) return;

    int cnt = review_counts[b];
    float item[NFAC];
    #pragma unroll
    for (int f = 0; f < NFAC; ++f) item[f] = 0.f;

    for (int r = 0; r < cnt; ++r) {
        const float* p = pooled + (size_t)(b * RPD + r) * NFAC;
        #pragma unroll
        for (int f = 0; f < NFAC; ++f) item[f] += p[f];
    }
    float denom = (float)(cnt > 0 ? cnt : 1);

    const float* wu = w_user + (size_t)user_indices[b] * NFAC;
    float acc = 0.f;
    #pragma unroll
    for (int f = 0; f < NFAC; ++f) acc += wu[f] * item[f];

    out[b] = acc / denom + bias[0];
}

extern "C" void kernel_launch(void* const* d_in, const int* in_sizes, int n_in,
                              void* d_out, int out_size, void* d_ws, size_t ws_size,
                              hipStream_t stream) {
    const int*   user_indices  = (const int*)  d_in[0];
    const int*   docs          = (const int*)  d_in[1];
    const int*   review_counts = (const int*)  d_in[2];
    const float* w_user        = (const float*)d_in[3];
    const float* embed         = (const float*)d_in[4];
    const float* convw         = (const float*)d_in[5];
    const float* bias          = (const float*)d_in[6];
    float*       out           = (float*)d_out;

    float*          pooled = (float*)d_ws;                                  // 160 KB
    unsigned short* wbf    = (unsigned short*)((char*)d_ws + POOLED_BYTES); // 40 KB

    convpmf_wconv_kernel<<<WGT_ELEMS / 256, 256, 0, stream>>>(convw, wbf);
    convpmf_review_kernel<<<BATCH * RPD, 256, 0, stream>>>(docs, embed, wbf, pooled);
    convpmf_reduce_kernel<<<1, 128, 0, stream>>>(pooled, user_indices, review_counts,
                                                 w_user, bias, out);
}
